// ControlledOUGenerator_66108136620297
// MI455X (gfx1250) — compile-verified, matrix operand order checked by reading, not run on hardware
//
#include <hip/hip_runtime.h>
#include <math.h>

// Problem dims (from reference): B=256, T=512, D=32, H=1024, O=1
#define B_DIM 256
#define T_DIM 512
#define D_DIM 32
#define H_DIM 1024

typedef __bf16 v16bf __attribute__((ext_vector_type(16)));
typedef float  v8f   __attribute__((ext_vector_type(8)));

// ---------------------------------------------------------------------------
// CDNA5 async copy: global -> LDS, 16 bytes, tracked by ASYNCcnt (no VGPR
// round trip). Inline asm (portable across toolchains). GV addressing mode:
// vdst = LDS byte address, vaddr = 64-bit global address, saddr = off.
// ---------------------------------------------------------------------------
__device__ __forceinline__ void async_copy_b128(unsigned lds_addr, const void* gsrc) {
    asm volatile("global_load_async_to_lds_b128 %0, %1, off"
                 :: "v"(lds_addr), "v"(gsrc)
                 : "memory");
}
__device__ __forceinline__ void wait_async0() {
    asm volatile("s_wait_asynccnt 0x0" ::: "memory");
}
// Scheduling fence: keep LDS fragment loads grouped ahead of the WMMA burst
// so the legalizer emits partial s_wait_dscnt instead of full drains.
__device__ __forceinline__ void sched_fence() {
    asm volatile("" ::: "memory");
}

// ---------------------------------------------------------------------------
// Per-step scalars: dt[i] = times[i+1]-times[i];  c[i] = sqrt(dt)*dW = eps*dt
// ---------------------------------------------------------------------------
__global__ void prep_scalars(const float* __restrict__ times,
                             const float* __restrict__ eps,
                             float* __restrict__ dts, float* __restrict__ cs) {
    int i = blockIdx.x * blockDim.x + threadIdx.x;
    if (i < T_DIM - 1) {
        float dt = times[i + 1] - times[i];
        float dw = eps[i] * sqrtf(dt);      // dW ~ N(0, sqrt(dt))
        dts[i] = dt;
        cs[i]  = sqrtf(dt) * dw;            // noise coefficient per reference
    }
}

// ---------------------------------------------------------------------------
// Pack W_theta|W_mu|W_sigma (fp32 row-major KxN, K=H, N=H) into bf16 WMMA
// B-fragment layout: per 32(K)x16(N) tile, lane l holds column n = n0+(l&15),
// K order {0..7,16..23} for lanes<16 and {8..15,24..31} for lanes>=16.
// Each 32x16 tile = 1024 contiguous bytes -> perfect for async B128 staging.
// ---------------------------------------------------------------------------
__global__ void pack_weights(const float* __restrict__ Wt,
                             const float* __restrict__ Wm,
                             const float* __restrict__ Ws,
                             __bf16* __restrict__ packed) {
    int e = blockIdx.x * blockDim.x + threadIdx.x;       // < 3*H*H
    int q      = e / (H_DIM * H_DIM);
    int rem    = e - q * (H_DIM * H_DIM);
    int tile   = rem >> 9;          // 512 elements per 32x16 tile
    int within = rem & 511;
    int lane   = within >> 4;
    int j      = within & 15;
    int ntile  = tile >> 5;         // tiles ordered n-major over 32 k-tiles
    int ktile  = tile & 31;
    int n      = (ntile << 4) + (lane & 15);
    int base   = (lane & 16) ? 8 : 0;
    int kloc   = (j < 8) ? (base + j) : (base + 16 + (j - 8));
    int k      = (ktile << 5) + kloc;
    const float* W = (q == 0) ? Wt : (q == 1) ? Wm : Ws;
    packed[e] = (__bf16)W[k * H_DIM + n];
}

// ---------------------------------------------------------------------------
// y0 = path[:,0,:] @ W_init + b_init  (tiny: 256x1024 outputs, K=32)
// ---------------------------------------------------------------------------
__global__ void init_state(const float* __restrict__ path,
                           const float* __restrict__ W_init,
                           const float* __restrict__ b_init,
                           float* __restrict__ y_f32, __bf16* __restrict__ y_bf16) {
    int idx = blockIdx.x * blockDim.x + threadIdx.x;     // < B*H
    int b = idx >> 10;
    int h = idx & (H_DIM - 1);
    const float* pr = path + (size_t)b * T_DIM * D_DIM;  // path[b,0,:]
    float acc = b_init[h];
#pragma unroll
    for (int d = 0; d < D_DIM; ++d) acc += pr[d] * W_init[d * H_DIM + h];
    y_f32[idx]  = acc;
    y_bf16[idx] = (__bf16)acc;
}

// ---------------------------------------------------------------------------
// decode: out[b, tcol] = dot(y[b,:], W_dec) + b_dec  — one wave32 per row,
// __shfl_xor butterfly reduction (deterministic, no atomics).
// ---------------------------------------------------------------------------
__global__ void decode_row(const float* __restrict__ y,
                           const float* __restrict__ W_dec,
                           const float* __restrict__ b_dec,
                           float* __restrict__ out, int tcol) {
    int wave = threadIdx.x >> 5;
    int lane = threadIdx.x & 31;
    int b = blockIdx.x * 8 + wave;
    const float* yr = y + (size_t)b * H_DIM;
    float p = 0.f;
#pragma unroll 8
    for (int h = lane; h < H_DIM; h += 32) p += yr[h] * W_dec[h];
#pragma unroll
    for (int off = 16; off >= 1; off >>= 1) p += __shfl_xor(p, off, 32);
    if (lane == 0) out[(size_t)b * T_DIM + tcol] = p + b_dec[0];
}

// ---------------------------------------------------------------------------
// Fused OU step, async-to-LDS software pipeline.
//   Workgroup = 8 waves: 4(M) x 2(H); WG tile = 64(M) x 64(H) x 3 matrices.
//   Stage = K=64 (2 k-tiles) = 32 KB LDS (24 KB B + 8 KB A), double buffered
//   (64 KB of the 320 KB/WGP). 16 stages; per stage: 1 s_wait_asynccnt +
//   1 barrier + 8 async B128 issues/thread + 12 WMMAs/wave.
//   Fragments for each k-subtile are batch-loaded before the WMMA burst.
// ---------------------------------------------------------------------------
#define STAGE_BYTES 32768
#define A_OFF       24576          // B: 24 chunks x 1 KB, then A: 8 x 1 KB

struct CopyPlan {
    const __bf16* bsrc[6];   // per-stage advance: +1024 elements
    unsigned      bdst[6];   // LDS offset within stage buffer
    const __bf16* asrc[2];   // per-stage advance: +64 elements
    unsigned      adst[2];
};

__device__ __forceinline__ void issue_stage(const CopyPlan& p, unsigned ldsBase,
                                            int stage) {
#pragma unroll
    for (int s = 0; s < 6; ++s)
        async_copy_b128(ldsBase + p.bdst[s], p.bsrc[s] + (size_t)stage * 1024);
#pragma unroll
    for (int s = 0; s < 2; ++s)
        async_copy_b128(ldsBase + p.adst[s], p.asrc[s] + (size_t)stage * 64);
}

__device__ __forceinline__ void compute_stage(const char* cb, int wm, int wn,
                                              int lane, v8f acc[3][2]) {
#pragma unroll
    for (int kt = 0; kt < 2; ++kt) {
        // batch-load this k-subtile's fragments: A + 6 B (56 VGPRs)
        v16bf a = *(const v16bf*)(cb + A_OFF + (kt * 4 + wm) * 1024 + lane * 32);
        v16bf b[6];
#pragma unroll
        for (int f = 0; f < 6; ++f) {
            int q = f >> 1, j = f & 1;
            b[f] = *(const v16bf*)(cb
                    + (((q * 4 + wn * 2 + j) * 2 + kt) * 1024) + lane * 32);
        }
        sched_fence();                 // loads stay grouped -> partial dscnt waits
#pragma unroll
        for (int f = 0; f < 6; ++f) {
            int q = f >> 1, j = f & 1;
            acc[q][j] = __builtin_amdgcn_wmma_f32_16x16x32_bf16(
                    false, a, false, b[f], (short)0, acc[q][j], false, false);
        }
    }
}

__global__ __launch_bounds__(256)
void ou_step(const float*  __restrict__ y_cur,
             const __bf16* __restrict__ yb_cur,
             const __bf16* __restrict__ packedW,
             const float*  __restrict__ b_theta,
             const float*  __restrict__ b_mu,
             const float*  __restrict__ b_sigma,
             const float*  __restrict__ dts,
             const float*  __restrict__ cs, int t,
             float*  __restrict__ y_nxt,
             __bf16* __restrict__ yb_nxt) {
    __shared__ __align__(128) char ldsbuf[2][STAGE_BYTES];

    const int tid  = threadIdx.x;
    const int lane = tid & 31;
    const int wave = tid >> 5;
    const int wm = wave & 3, wn = wave >> 2;
    const int m0wg = blockIdx.y * 64;
    const int h0wg = blockIdx.x * 64;

    const unsigned base0 = (unsigned)(unsigned long long)(&ldsbuf[0][0]);
    const unsigned base1 = (unsigned)(unsigned long long)(&ldsbuf[1][0]);
    const char* lds0 = &ldsbuf[0][0];
    const char* lds1 = &ldsbuf[1][0];

    // ---- per-thread async-copy plan (constant across stages) ----
    CopyPlan plan;
#pragma unroll
    for (int s = 0; s < 6; ++s) {            // B: 1536 x 16B per stage, 6/thread
        int c     = tid + s * 256;           // 0..1535
        int chunk = c >> 6;                  // 0..23 = (q*4+nt)*2 + ktSub
        int i     = c & 63;                  // 16B unit within 1 KB chunk
        int pair  = chunk >> 1;              // q*4 + nt
        int ktSub = chunk & 1;
        int q     = pair >> 2;
        int nt    = pair & 3;
        plan.bsrc[s] = packedW + (size_t)q * H_DIM * H_DIM
                     + ((size_t)(((h0wg >> 4) + nt) * 32 + ktSub) << 9)
                     + (size_t)i * 8;
        plan.bdst[s] = chunk * 1024 + i * 16;
    }
#pragma unroll
    for (int s = 0; s < 2; ++s) {            // A: 512 x 16B per stage, 2/thread
        int idx   = tid + s * 256;           // 0..511
        int ktSub = idx >> 8;
        int rem   = idx & 255;
        int mtile = rem >> 6;
        int ln    = (rem >> 1) & 31;
        int half  = rem & 1;
        int mrow  = m0wg + mtile * 16 + (ln & 15);
        int kloc  = ((ln & 16) ? 8 : 0) + half * 16;
        plan.asrc[s] = yb_cur + (size_t)mrow * H_DIM + ktSub * 32 + kloc;
        plan.adst[s] = A_OFF + (ktSub * 4 + mtile) * 1024 + ln * 32 + half * 16;
    }

    v8f acc[3][2];
#pragma unroll
    for (int q = 0; q < 3; ++q)
#pragma unroll
        for (int j = 0; j < 2; ++j) { v8f z = {}; acc[q][j] = z; }

    issue_stage(plan, base0, 0);

#pragma unroll 1
    for (int it = 0; it < 8; ++it) {         // 16 stages, two bodies per iter
        const int s0 = 2 * it;
        // ---- body A: stage s0 in buf0 ----
        wait_async0();
        __syncthreads();                     // stage s0 visible WG-wide; also
                                             // proves buf1 readers (s0-1) done
        issue_stage(plan, base1, s0 + 1);    // s0+1 <= 15 always
        compute_stage(lds0, wm, wn, lane, acc);
        // ---- body B: stage s0+1 in buf1 ----
        wait_async0();
        __syncthreads();
        if (s0 + 2 < 16) issue_stage(plan, base0, s0 + 2);
        compute_stage(lds1, wm, wn, lane, acc);
    }

    // ---- fused OU update epilogue ----
    const float dt = dts[t];
    const float c  = cs[t];
    const int m0    = m0wg + wm * 16;
    const int h0    = h0wg + wn * 32;
    const int mbase = m0 + ((lane & 16) ? 8 : 0);   // C/D layout: lanes>=16 -> M+8
#pragma unroll
    for (int j = 0; j < 2; ++j) {
        const int h = h0 + j * 16 + (lane & 15);
        const float bt = b_theta[h], bm = b_mu[h], bs = b_sigma[h];
#pragma unroll
        for (int r = 0; r < 8; ++r) {
            const int m = mbase + r;
            const size_t idx = (size_t)m * H_DIM + h;
            const float yo = y_cur[idx];
            const float th = acc[0][j][r] + bt;
            const float mu = acc[1][j][r] + bm;
            const float sg = acc[2][j][r] + bs;
            const float yn = yo + th * (mu - yo) * dt + sg * c;
            y_nxt[idx]  = yn;
            yb_nxt[idx] = (__bf16)yn;
        }
    }
}

// ---------------------------------------------------------------------------
extern "C" void kernel_launch(void* const* d_in, const int* in_sizes, int n_in,
                              void* d_out, int out_size, void* d_ws, size_t ws_size,
                              hipStream_t stream) {
    (void)in_sizes; (void)n_in; (void)out_size; (void)ws_size;

    const float* path    = (const float*)d_in[0];
    const float* times   = (const float*)d_in[1];
    const float* eps     = (const float*)d_in[2];
    const float* W_init  = (const float*)d_in[3];
    const float* b_init  = (const float*)d_in[4];
    const float* W_theta = (const float*)d_in[5];
    const float* b_theta = (const float*)d_in[6];
    const float* W_mu    = (const float*)d_in[7];
    const float* b_mu    = (const float*)d_in[8];
    const float* W_sigma = (const float*)d_in[9];
    const float* b_sigma = (const float*)d_in[10];
    const float* W_dec   = (const float*)d_in[11];
    const float* b_dec   = (const float*)d_in[12];
    float* out = (float*)d_out;

    // Workspace layout (~9 MB): fp32 y ping/pong, bf16 y ping/pong,
    // packed bf16 weights (6 MB), per-step scalars.
    char* ws = (char*)d_ws;
    float*  yA     = (float*) (ws + 0);
    float*  yB     = (float*) (ws + (1u << 20));
    __bf16* ybA    = (__bf16*)(ws + (2u << 20));
    __bf16* ybB    = (__bf16*)(ws + (2u << 20) + (512u << 10));
    __bf16* packed = (__bf16*)(ws + (3u << 20));
    float*  dts    = (float*) (ws + (9u << 20));
    float*  cs     = dts + 1024;

    prep_scalars<<<2, 256, 0, stream>>>(times, eps, dts, cs);
    pack_weights<<<(3 * H_DIM * H_DIM) / 256, 256, 0, stream>>>(W_theta, W_mu, W_sigma, packed);
    init_state<<<(B_DIM * H_DIM) / 256, 256, 0, stream>>>(path, W_init, b_init, yA, ybA);
    decode_row<<<B_DIM / 8, 256, 0, stream>>>(yA, W_dec, b_dec, out, 0);

    float*  yc  = yA;  float*  yn  = yB;
    __bf16* ybc = ybA; __bf16* ybn = ybB;
    dim3 grid(H_DIM / 64, B_DIM / 64);
    for (int t = 0; t < T_DIM - 1; ++t) {
        ou_step<<<grid, 256, 0, stream>>>(yc, ybc, packed, b_theta, b_mu, b_sigma,
                                          dts, cs, t, yn, ybn);
        decode_row<<<B_DIM / 8, 256, 0, stream>>>(yn, W_dec, b_dec, out, t + 1);
        float*  tf = yc;  yc  = yn;  yn  = tf;
        __bf16* tb = ybc; ybc = ybn; ybn = tb;
    }
}
